// Temporal_self_att_27994596835628
// MI455X (gfx1250) — matrix operand order repaired and checked by reading, not compiled, not run on hardware
//
#include <hip/hip_runtime.h>

typedef __attribute__((ext_vector_type(16))) _Float16 v16h;
typedef __attribute__((ext_vector_type(8)))  float    v8f;

union AF { v16h v; uint4 q[2]; unsigned int u[8]; _Float16 h[16]; };
union CF { v8f v; float f[8]; };
union PK { uint4 q; _Float16 h[8]; };

#define WMMA_F32_F16(a,b,c) \
  __builtin_amdgcn_wmma_f32_16x16x32_f16(false,(a),false,(b),(short)0,(c),false,false)

// A-matrix fragment (16x32 f16), row-major source, leading dim ld (ld*2 must be 16B mult).
// Lane m=L&15, hi=L>>4; halves [0..7] = A[m][hi*8..+7], [8..15] = A[m][16+hi*8..+7].
__device__ __forceinline__ v16h ldA16(const _Float16* p, int ld, int lane){
  const int m = lane & 15, hi = lane >> 4;
  AF f;
  f.q[0] = *(const uint4*)(p + m*ld + hi*8);
  f.q[1] = *(const uint4*)(p + m*ld + hi*8 + 16);
  return f.v;
}

// B-matrix fragment (32x16 f16) where B = M^T, M row-major (B[k][n] = M[n][k]).
// Lane n=L&15, hi=L>>4; halves [0..7] = M[n][hi*8..+7], [8..15] = M[n][16+hi*8..+7] -> two b128 loads.
__device__ __forceinline__ v16h ldBt16(const _Float16* p, int ld, int lane){
  const int n = lane & 15, hi = lane >> 4;
  AF f;
  f.q[0] = *(const uint4*)(p + n*ld + hi*8);
  f.q[1] = *(const uint4*)(p + n*ld + hi*8 + 16);
  return f.v;
}

// ---------------------------------------------------------------- kernel 0: f32 weights -> f16 TRANSPOSED
__global__ __launch_bounds__(256) void cvtw_kernel(
    const float* __restrict__ Wq, const float* __restrict__ Wk,
    const float* __restrict__ Wv, const float* __restrict__ Wo,
    _Float16* __restrict__ WqT, _Float16* __restrict__ WkT,
    _Float16* __restrict__ WvT, _Float16* __restrict__ WoT){
  const int i = blockIdx.x*256 + threadIdx.x;
  if (i < 64*256){
    const int k = i >> 8, n = i & 255;        // W[k][n], 64x256
    WqT[n*64 + k] = (_Float16)Wq[i];
    WkT[n*64 + k] = (_Float16)Wk[i];
    WvT[n*64 + k] = (_Float16)Wv[i];
  }
  if (i < 256*256){
    const int k = i >> 8, n = i & 255;        // Wo[k][n], 256x256
    WoT[n*256 + k] = (_Float16)Wo[i];
  }
}

// ---------------------------------------------------------------- kernel 1: QKV projection
// grid (8 mblocks, 128 bn), 256 threads = 8 waves; wave: 16 t-rows x 128 cols.
// Q,K stored [bn][t][256] f16 ; V stored TRANSPOSED [bn][256][512] f16.
__global__ __launch_bounds__(256) void qkv_kernel(
    const float* __restrict__ x,
    const _Float16* __restrict__ WqT, const _Float16* __restrict__ WkT, const _Float16* __restrict__ WvT,
    const float* __restrict__ bq, const float* __restrict__ bk, const float* __restrict__ bv,
    _Float16* __restrict__ Qo, _Float16* __restrict__ Ko, _Float16* __restrict__ Vt){
  __shared__ __align__(16) _Float16 xsA[64][72];   // A tile: [t_local][f], padded (144B rows)
  const int mb = blockIdx.x, bn = blockIdx.y;
  const int tid = threadIdx.x, wave = tid >> 5, lane = tid & 31;
  const float* xp = x + (size_t)bn*64*512;         // x[bn][f][t]
#pragma unroll
  for (int i = 0; i < 16; ++i){
    const int idx = tid + i*256;
    const int f = idx >> 6, tl = idx & 63;
    xsA[tl][f] = (_Float16)xp[f*512 + mb*64 + tl]; // transpose on the fly
  }
  __syncthreads();
  const int wm = wave >> 1, wn = wave & 1;
  const int n = lane & 15, hi = lane >> 4;
  const v16h a0 = ldA16(&xsA[wm*16][0],  72, lane);
  const v16h a1 = ldA16(&xsA[wm*16][32], 72, lane);
  const _Float16* Ws[3] = {WqT, WkT, WvT};
  const float*    bs[3] = {bq, bk, bv};
  _Float16*       Os[2] = {Qo, Ko};
#pragma unroll
  for (int w = 0; w < 3; ++w){
#pragma unroll 2
    for (int nt = 0; nt < 8; ++nt){
      const int n0 = wn*128 + nt*16;
      v16h b0 = ldBt16(Ws[w] + n0*64,      64, lane);   // W[k=0..31][n0+n]
      v16h b1 = ldBt16(Ws[w] + n0*64 + 32, 64, lane);   // W[k=32..63][n0+n]
      v8f c = {};
      c = WMMA_F32_F16(a0, b0, c);
      c = WMMA_F32_F16(a1, b1, c);
      const float bb = bs[w][n0 + n];
      CF cf; cf.v = c;
      const int t0 = mb*64 + wm*16 + hi*8;
      if (w < 2){
        _Float16* O = Os[w] + (size_t)bn*512*256;
#pragma unroll
        for (int j = 0; j < 8; ++j)
          O[(t0 + j)*256 + n0 + n] = (_Float16)(cf.f[j] + bb);
      } else {
        // V transposed: Vt[bn][ch][t]; 8 consecutive t per lane -> one b128 store
        PK pk;
#pragma unroll
        for (int j = 0; j < 8; ++j) pk.h[j] = (_Float16)(cf.f[j] + bb);
        *(uint4*)(Vt + (size_t)bn*256*512 + (size_t)(n0 + n)*512 + t0) = pk.q;
      }
    }
  }
}

// ---------------------------------------------------------------- kernel 2: attention
// grid (16 qblocks of 32 rows, 4 heads, 128 bn), 256 threads.
// Dynamic LDS: S f32 [32][520] (66560 B) + P f16 [32][520] (33280 B) = 99840 B
//   -> 3 workgroups per WGP (320 KB), so softmax/WMMA phases of different WGs overlap.
__global__ __launch_bounds__(256) void attn_kernel(
    const _Float16* __restrict__ Q, const _Float16* __restrict__ K,
    const _Float16* __restrict__ Vt, _Float16* __restrict__ AO){
  extern __shared__ char smem[];
  float*    S = (float*)smem;                    // [32][520]
  _Float16* P = (_Float16*)(smem + 32*520*4);    // [32][520]
  __shared__ float red[32][8];
  const int qb = blockIdx.x, h = blockIdx.y, bn = blockIdx.z;
  const int tid = threadIdx.x, wave = tid >> 5, lane = tid & 31;
  const int n = lane & 15, hi = lane >> 4;
  const _Float16* Qh  = Q  + (size_t)bn*512*256 + h*64;
  const _Float16* Kh  = K  + (size_t)bn*512*256 + h*64;
  const _Float16* Vth = Vt + (size_t)bn*256*512 + (size_t)h*64*512;

  // Phase A: S = scale * Q K^T   (2 m-tiles x 32 n-tiles; wave: 1 m-tile x 8 n-tiles)
  {
    const int mt = wave & 1, nh = wave >> 1;
    const v16h a0 = ldA16(Qh + (qb*32 + mt*16)*256,      256, lane);
    const v16h a1 = ldA16(Qh + (qb*32 + mt*16)*256 + 32, 256, lane);
#pragma unroll 2
    for (int nt2 = 0; nt2 < 8; ++nt2){
      const int key0 = (nh*8 + nt2)*16;
      v16h b0 = ldBt16(Kh + key0*256,      256, lane);
      v16h b1 = ldBt16(Kh + key0*256 + 32, 256, lane);
      v8f c = {};
      c = WMMA_F32_F16(a0, b0, c);
      c = WMMA_F32_F16(a1, b1, c);
      CF cf; cf.v = c;
#pragma unroll
      for (int j = 0; j < 8; ++j)
        S[(mt*16 + hi*8 + j)*520 + key0 + n] = cf.f[j] * 0.125f;
    }
  }
  __syncthreads();

  // Phase B: row softmax, 8 threads per row with LDS partial reduce
  {
    const int row = tid >> 3, qc = tid & 7;
    const int c0 = qc*64;
    float mx = -1e30f;
    for (int c2 = 0; c2 < 64; ++c2) mx = fmaxf(mx, S[row*520 + c0 + c2]);
    red[row][qc] = mx;
    __syncthreads();
#pragma unroll
    for (int i = 0; i < 8; ++i) mx = fmaxf(mx, red[row][i]);
    __syncthreads();
    float sum = 0.f;
    for (int c2 = 0; c2 < 64; ++c2){
      const float e = __expf(S[row*520 + c0 + c2] - mx);
      sum += e;
      S[row*520 + c0 + c2] = e;
    }
    red[row][qc] = sum;
    __syncthreads();
    sum = 0.f;
#pragma unroll
    for (int i = 0; i < 8; ++i) sum += red[row][i];
    const float inv = 1.f / sum;
    for (int c2 = 0; c2 < 64; ++c2)
      P[row*520 + c0 + c2] = (_Float16)(S[row*520 + c0 + c2] * inv);
  }
  __syncthreads();

  // Phase C: O = P @ V  (2 m-tiles x 4 n-tiles; one tile per wave, 16 k-steps,
  //  even/odd k-step accumulators to keep two independent WMMA chains in flight)
  {
    const int mt = wave & 1, nt = wave >> 1;
    v8f c0 = {}, c1 = {};
#pragma unroll 2
    for (int kk = 0; kk < 16; kk += 2){
      v16h a0 = ldA16(P + mt*16*520 + kk*32,      520, lane);
      v16h a1 = ldA16(P + mt*16*520 + kk*32 + 32, 520, lane);
      v16h b0 = ldBt16(Vth + (size_t)nt*16*512 + kk*32,      512, lane);
      v16h b1 = ldBt16(Vth + (size_t)nt*16*512 + kk*32 + 32, 512, lane);
      c0 = WMMA_F32_F16(a0, b0, c0);
      c1 = WMMA_F32_F16(a1, b1, c1);
    }
    _Float16* AOp = AO + (size_t)bn*512*256 + h*64;
    CF f0, f1; f0.v = c0; f1.v = c1;
#pragma unroll
    for (int j = 0; j < 8; ++j){
      const int t = qb*32 + mt*16 + hi*8 + j;
      AOp[t*256 + nt*16 + n] = (_Float16)(f0.f[j] + f1.f[j]);
    }
  }
}

// ---------------------------------------------------------------- kernel 3: output projection + transpose
// grid (8 mblocks, 128 bn); wave: 16 t-rows x 128 channels (8 independent accumulators).
__global__ __launch_bounds__(256) void oproj_kernel(
    const _Float16* __restrict__ AO, const _Float16* __restrict__ WoT,
    const float* __restrict__ bo, float* __restrict__ out){
  const int mb = blockIdx.x, bn = blockIdx.y;
  const int tid = threadIdx.x, wave = tid >> 5, lane = tid & 31;
  const int wm = wave >> 1, wn = wave & 1;
  const int n = lane & 15, hi = lane >> 4;
  const _Float16* Ap = AO + (size_t)bn*512*256 + (mb*64 + wm*16)*256;
  v8f zero = {};
  v8f acc[8];
#pragma unroll
  for (int i = 0; i < 8; ++i) acc[i] = zero;
#pragma unroll 1
  for (int kk = 0; kk < 16; ++kk){
    v16h a = ldA16(Ap + kk*32, 256, lane);
#pragma unroll
    for (int nt = 0; nt < 8; ++nt){
      v16h b = ldBt16(WoT + (wn*128 + nt*16)*256 + kk*32, 256, lane);
      acc[nt] = WMMA_F32_F16(a, b, acc[nt]);
    }
  }
  float* op = out + (size_t)bn*256*512;
  const int t0 = mb*64 + wm*16 + hi*8;    // 8 consecutive t per lane -> b128 stores
#pragma unroll
  for (int nt = 0; nt < 8; ++nt){
    const int ch = wn*128 + nt*16 + n;
    const float bb = bo[ch];
    CF cf; cf.v = acc[nt];
    float4 v0 = make_float4(cf.f[0]+bb, cf.f[1]+bb, cf.f[2]+bb, cf.f[3]+bb);
    float4 v1 = make_float4(cf.f[4]+bb, cf.f[5]+bb, cf.f[6]+bb, cf.f[7]+bb);
    *(float4*)(op + (size_t)ch*512 + t0)     = v0;
    *(float4*)(op + (size_t)ch*512 + t0 + 4) = v1;
  }
}

// ---------------------------------------------------------------- launch
extern "C" void kernel_launch(void* const* d_in, const int* in_sizes, int n_in,
                              void* d_out, int out_size, void* d_ws, size_t ws_size,
                              hipStream_t stream){
  (void)in_sizes; (void)n_in; (void)out_size; (void)ws_size;
  const float* x  = (const float*)d_in[0];
  const float* Wq = (const float*)d_in[1];
  const float* bq = (const float*)d_in[2];
  const float* Wk = (const float*)d_in[3];
  const float* bk = (const float*)d_in[4];
  const float* Wv = (const float*)d_in[5];
  const float* bv = (const float*)d_in[6];
  const float* Wo = (const float*)d_in[7];
  const float* bo = (const float*)d_in[8];
  float* out = (float*)d_out;

  char* ws = (char*)d_ws;
  _Float16* WqT = (_Float16*)(ws);
  _Float16* WkT = (_Float16*)(ws + 32768);
  _Float16* WvT = (_Float16*)(ws + 65536);
  _Float16* WoT = (_Float16*)(ws + 98304);
  const size_t qkvBytes = (size_t)128*512*256*2;   // 32 MiB each
  _Float16* Qb  = (_Float16*)(ws + 262144);
  _Float16* Kb  = (_Float16*)(ws + 262144 + qkvBytes);
  _Float16* Vtb = (_Float16*)(ws + 262144 + 2*qkvBytes);
  _Float16* AOb = (_Float16*)(ws + 262144 + 3*qkvBytes);

  cvtw_kernel<<<256, 256, 0, stream>>>(Wq, Wk, Wv, Wo, WqT, WkT, WvT, WoT);
  qkv_kernel<<<dim3(8,128), 256, 0, stream>>>(x, WqT, WkT, WvT, bq, bk, bv, Qb, Kb, Vtb);
  attn_kernel<<<dim3(16,4,128), 256, 99840, stream>>>(Qb, Kb, Vtb, AOb);
  oproj_kernel<<<dim3(8,128), 256, 0, stream>>>(AOb, WoT, bo, out);
}
